// my_Net_65154653880788
// MI455X (gfx1250) — 8-bit matrix operand order repaired and checked by reading, compiled, not run on hardware
//
#include <hip/hip_runtime.h>

typedef __attribute__((ext_vector_type(8))) int v8i;
typedef unsigned int u32;

#define CI     16
#define CO     32
#define HIN    224
#define WIN    224
#define HOUT   222
#define WOUT   222
#define KH     3
#define KW     3
#define TILE_W 64
#define COLS   66
#define PADW   20

__device__ __forceinline__ int q16(float x) {
    float v = floorf(x * 4096.0f);
    v = fminf(fmaxf(v, -32768.0f), 32767.0f);
    return (int)v;
}

__global__ void pack_weights_kernel(const float* __restrict__ w,
                                    u32* __restrict__ ws) {
    int idx = blockIdx.x * blockDim.x + threadIdx.x;
    if (idx >= KH * CI * CO) return;
    int co = idx & (CO - 1);
    int ci = (idx >> 5) & (CI - 1);
    int r  = idx >> 9;
    const float* wp = w + ((co * CI + ci) * KH + r) * KW;
    int q0 = q16(wp[0]);
    int q1 = q16(wp[1]);
    int q2 = q16(wp[2]);
    u32 hi = (u32)((q0 >> 8) & 0xFF)
           | ((u32)((q1 >> 8) & 0xFF) << 8)
           | ((u32)((q2 >> 8) & 0xFF) << 16);
    u32 lo = (u32)(q0 & 0xFF)
           | ((u32)(q1 & 0xFF) << 8)
           | ((u32)(q2 & 0xFF) << 16);
    ws[((0 * KH + r) * CO + co) * CI + ci] = hi;
    ws[((1 * KH + r) * CO + co) * CI + ci] = lo;
}

__global__ __launch_bounds__(128)
void conv_wmma_kernel(const float* __restrict__ x,
                      const u32* __restrict__ wpk,
                      float* __restrict__ out) {
    __shared__ __align__(16) unsigned char ldsB[2][KH][CI][72];
    __shared__ __align__(16) u32 ldsA[2][KH][TILE_W][PADW];
    __shared__ __align__(16) u32 ldsW[2][KH][CO][PADW];

    const int wt  = blockIdx.x;
    const int h   = blockIdx.y;
    const int b   = blockIdx.z;
    const int wt0 = wt * TILE_W;
    const int tid = threadIdx.x;

    for (int idx = tid; idx < KH * CI * COLS; idx += 128) {
        int c  = idx % COLS;
        int ci = (idx / COLS) % CI;
        int r  = idx / (COLS * CI);
        int g  = wt0 + c;
        int X  = 0;
        if (g < WIN) X = q16(x[((b * CI + ci) * HIN + (h + r)) * WIN + g]);
        ldsB[0][r][ci][c] = (unsigned char)((X >> 8) & 0xFF);
        ldsB[1][r][ci][c] = (unsigned char)(X & 0xFF);
    }
    for (int idx = tid; idx < 2 * KH * CO * CI; idx += 128) {
        int ci  = idx & (CI - 1);
        int row = idx >> 4;
        (&ldsW[0][0][0][0])[row * PADW + ci] = wpk[idx];
    }
    __syncthreads();

    for (int u = tid; u < 2 * KH * CI * 4; u += 128) {
        int ci = u & 15;
        int rb = (u >> 4) & 3;
        int pr = u >> 6;
        int r  = pr % KH;
        int p  = pr / KH;
        int c0 = rb * 16;
        const u32* row32 = (const u32*)&ldsB[p][r][ci][0];
        u32 d[5];
        #pragma unroll
        for (int k = 0; k < 5; ++k) d[k] = row32[(c0 >> 2) + k];
        #pragma unroll
        for (int j = 0; j < 16; ++j) {
            int k  = j >> 2;
            int sh = (j & 3) * 8;
            unsigned long long pw =
                ((unsigned long long)d[k + 1] << 32) | (unsigned long long)d[k];
            ldsA[p][r][c0 + j][ci] = (u32)((pw >> sh) & 0xFFFFFFu);
        }
    }
    __syncthreads();

    const int lane = tid & 31;
    const int wave = tid >> 5;
    const int half = (lane >> 4) & 1;
    const int m    = lane & 15;
    const int col  = wave * 16 + m;

    v8i acc[2][4];
    #pragma unroll
    for (int t = 0; t < 2; ++t)
        #pragma unroll
        for (int s = 0; s < 4; ++s)
            #pragma unroll
            for (int e = 0; e < 8; ++e) acc[t][s][e] = 0;

    #pragma unroll
    for (int r = 0; r < KH; ++r) {
        v8i a_hi, a_lo;
        {
            const u32* aH = &ldsA[0][r][col][0];
            const u32* aL = &ldsA[1][r][col][0];
            #pragma unroll
            for (int pz = 0; pz < 4; ++pz) {
                int cb = 8 * (pz >> 1) + 4 * half + 2 * (pz & 1);
                uint2 dh = *(const uint2*)(aH + cb);
                uint2 dl = *(const uint2*)(aL + cb);
                a_hi[2 * pz]     = (int)dh.x;
                a_hi[2 * pz + 1] = (int)dh.y;
                a_lo[2 * pz]     = (int)dl.x;
                a_lo[2 * pz + 1] = (int)dl.y;
            }
        }
        #pragma unroll
        for (int t = 0; t < 2; ++t) {
            const u32* wH = &ldsW[0][r][t * 16 + m][0];
            const u32* wL = &ldsW[1][r][t * 16 + m][0];
            uint4 h0 = *(const uint4*)(wH + 4 * half);
            uint4 h1 = *(const uint4*)(wH + 8 + 4 * half);
            uint4 l0 = *(const uint4*)(wL + 4 * half);
            uint4 l1 = *(const uint4*)(wL + 8 + 4 * half);
            v8i b_hi, b_lo;
            b_hi[0] = (int)h0.x; b_hi[1] = (int)h0.y; b_hi[2] = (int)h0.z; b_hi[3] = (int)h0.w;
            b_hi[4] = (int)h1.x; b_hi[5] = (int)h1.y; b_hi[6] = (int)h1.z; b_hi[7] = (int)h1.w;
            b_lo[0] = (int)l0.x; b_lo[1] = (int)l0.y; b_lo[2] = (int)l0.z; b_lo[3] = (int)l0.w;
            b_lo[4] = (int)l1.x; b_lo[5] = (int)l1.y; b_lo[6] = (int)l1.z; b_lo[7] = (int)l1.w;
            acc[t][0] = __builtin_amdgcn_wmma_i32_16x16x64_iu8(true,  a_hi, true,  b_hi, acc[t][0], false, false);
            acc[t][1] = __builtin_amdgcn_wmma_i32_16x16x64_iu8(true,  a_hi, false, b_lo, acc[t][1], false, false);
            acc[t][2] = __builtin_amdgcn_wmma_i32_16x16x64_iu8(false, a_lo, true,  b_hi, acc[t][2], false, false);
            acc[t][3] = __builtin_amdgcn_wmma_i32_16x16x64_iu8(false, a_lo, false, b_lo, acc[t][3], false, false);
        }
    }

    #pragma unroll
    for (int t = 0; t < 2; ++t) {
        int co = t * 16 + m;
        #pragma unroll
        for (int rr = 0; rr < 8; ++rr) {
            int pix = wt0 + wave * 16 + rr + 8 * half;
            long long S = ((long long)acc[t][0][rr] << 16)
                        + ((long long)((long long)acc[t][1][rr] + (long long)acc[t][2][rr]) << 8)
                        +  (long long)acc[t][3][rr];
            long long q = S >> 4;
            q = q < -2147483648LL ? -2147483648LL
                                  : (q > 2147483647LL ? 2147483647LL : q);
            if (pix < WOUT) {
                out[((b * CO + co) * HOUT + h) * WOUT + pix] =
                    (float)q * (1.0f / 1048576.0f);
            }
        }
    }
}

extern "C" void kernel_launch(void* const* d_in, const int* in_sizes, int n_in,
                              void* d_out, int out_size, void* d_ws, size_t ws_size,
                              hipStream_t stream) {
    const float* x = (const float*)d_in[0];
    const float* w = (const float*)d_in[1];
    float* out = (float*)d_out;
    u32* wpk = (u32*)d_ws;

    pack_weights_kernel<<<(KH * CI * CO + 255) / 256, 256, 0, stream>>>(w, wpk);

    dim3 grid((WOUT + TILE_W - 1) / TILE_W, HOUT, 16);
    conv_wmma_kernel<<<grid, 128, 0, stream>>>(x, wpk, out);
}
